// Block_31645319037454
// MI455X (gfx1250) — matrix operand order repaired and checked by reading, not compiled, not run on hardware
//
#include <hip/hip_runtime.h>
#include <hip/hip_bf16.h>

// ---------------------------------------------------------------------------
// Types for CDNA5 WMMA (wave32, v_wmma_f32_16x16x32_bf16)
// ---------------------------------------------------------------------------
typedef __bf16 bf16_t;
typedef __attribute__((ext_vector_type(16))) __bf16 v16bf;
typedef __attribute__((ext_vector_type(8)))  float  v8f;

union Frag {
    v16bf v;
    uint4 q[2];
};

#define WMMA_BF16(a, b, c) \
    __builtin_amdgcn_wmma_f32_16x16x32_bf16(false, (a), false, (b), (short)0, (c), false, false)

// CDNA5 async memory->LDS copy (ASYNCcnt-tracked, no VGPR staging).
// vdst = LDS byte address (low 32 bits of generic pointer), vaddr = 64-bit
// global address. Portable across toolchains via inline asm (see bridge doc).
__device__ __forceinline__ void async_copy16(const bf16_t* g, bf16_t* l)
{
    unsigned long long ga = (unsigned long long)g;
    unsigned la = (unsigned)(unsigned long long)l;
    asm volatile("global_load_async_to_lds_b128 %0, %1, off"
                 :: "v"(la), "v"(ga)
                 : "memory");
}
__device__ __forceinline__ void wait_async_all()
{
    asm volatile("s_wait_asynccnt 0" ::: "memory");
}

// ---------------------------------------------------------------------------
// Problem constants
// ---------------------------------------------------------------------------
static constexpr int BB = 4;
static constexpr int TT = 2048;
static constexpr int CC = 1024;
static constexpr int HH = 16;
static constexpr int HS = 64;
static constexpr int MM = BB * TT;        // 8192 rows

// ---------------------------------------------------------------------------
// Weight convert + transpose: in fp32 [batch][K][N] -> out bf16 [batch][N][K]
// ---------------------------------------------------------------------------
__global__ void transpose_to_bf16_kernel(const float* __restrict__ in,
                                         bf16_t* __restrict__ out,
                                         int batch, int K, int N)
{
    size_t idx   = (size_t)blockIdx.x * blockDim.x + threadIdx.x;
    size_t total = (size_t)batch * K * N;
    if (idx >= total) return;
    int n = (int)(idx % N);
    size_t t = idx / N;
    int k = (int)(t % K);
    int b = (int)(t / K);
    out[((size_t)b * N + n) * K + k] = (bf16_t)in[idx];
}

// ---------------------------------------------------------------------------
// LayerNorm (C == 1024, 256 threads, 4 elems/thread) -> bf16
// ---------------------------------------------------------------------------
__global__ __launch_bounds__(256)
void layernorm_bf16_kernel(const float* __restrict__ x,
                           const float* __restrict__ g,
                           const float* __restrict__ beta,
                           bf16_t* __restrict__ out)
{
    const int row = blockIdx.x;
    const float* xr = x + (size_t)row * CC;
    const int tid = threadIdx.x;

    float vals[4];
    float s = 0.f, ss = 0.f;
    for (int i = 0; i < 4; ++i) {
        float v = xr[tid + i * 256];
        vals[i] = v;
        s += v;
        ss += v * v;
    }
    for (int off = 16; off > 0; off >>= 1) {
        s  += __shfl_xor(s,  off, 32);
        ss += __shfl_xor(ss, off, 32);
    }
    __shared__ float red0[8], red1[8];
    const int wv = tid >> 5, lane = tid & 31;
    if (lane == 0) { red0[wv] = s; red1[wv] = ss; }
    __syncthreads();
    if (wv == 0) {
        float a = (lane < 8) ? red0[lane] : 0.f;
        float c = (lane < 8) ? red1[lane] : 0.f;
        for (int off = 4; off > 0; off >>= 1) {
            a += __shfl_xor(a, off, 32);
            c += __shfl_xor(c, off, 32);
        }
        if (lane == 0) { red0[0] = a; red1[0] = c; }
    }
    __syncthreads();
    const float mu   = red0[0] * (1.0f / CC);
    const float var  = red1[0] * (1.0f / CC) - mu * mu;
    const float rstd = rsqrtf(var + 1e-5f);

    bf16_t* orow = out + (size_t)row * CC;
    for (int i = 0; i < 4; ++i) {
        int c = tid + i * 256;
        orow[c] = (bf16_t)((vals[i] - mu) * rstd * g[c] + beta[c]);
    }
}

// ---------------------------------------------------------------------------
// WMMA GEMM: C[M,N] = A[M,K](bf16 rowmajor) * BT[N,K](bf16, pre-transposed)
// Block tile 64x128, BK=32, 256 threads = 8 waves (2x4), each wave 32x32.
// Double-buffered LDS staging via global_load_async_to_lds_b128.
// MODE 0: out bf16 rowmajor [M,N]            (optional ReLU)
// MODE 1: out bf16 scatter Q/K -> [B,H,T,HS]
// MODE 2: out bf16 scatter V^T -> [B,H,HS,T]
// MODE 3: out f32 rowmajor + bias + residual
// ---------------------------------------------------------------------------
template <int MODE, bool RELU>
__global__ __launch_bounds__(256)
void gemm_wmma_kernel(const bf16_t* __restrict__ A,
                      const bf16_t* __restrict__ BT,
                      const float*  __restrict__ bias,
                      const float*  __restrict__ res,
                      void* __restrict__ outp,
                      int M, int N, int K)
{
    constexpr int BM = 64, BN = 128, BK = 32;
    __shared__ bf16_t sA[2][BM * BK];   // [m][k]
    __shared__ bf16_t sB[2][BN * BK];   // [n][k]

    const int tid  = threadIdx.x;
    const int lane = tid & 31;
    const int wv   = tid >> 5;
    const int wm   = wv >> 2;        // 0..1
    const int wn   = wv & 3;         // 0..3
    const int lm   = lane & 15;
    const int lh   = lane >> 4;
    const int m0b  = blockIdx.y * BM;
    const int n0b  = blockIdx.x * BN;

    // per-thread staging coordinates
    const int ar  = tid >> 2;             // A row 0..63
    const int ac  = (tid & 3) << 3;       // A col 0,8,16,24
    const int br  = tid >> 1;             // B row 0..127
    const int bc  = (tid & 1) << 4;       // B col 0,16

    auto stage = [&](int buf, int k0) {
        async_copy16(&A[(size_t)(m0b + ar) * K + k0 + ac], &sA[buf][ar * BK + ac]);
        const bf16_t* src = &BT[(size_t)(n0b + br) * K + k0 + bc];
        async_copy16(src,     &sB[buf][br * BK + bc]);
        async_copy16(src + 8, &sB[buf][br * BK + bc + 8]);
    };

    v8f acc[2][2] = {};

    stage(0, 0);
    wait_async_all();
    __syncthreads();

    int cur = 0;
    for (int k0 = 0; k0 < K; k0 += BK) {
        const bool more = (k0 + BK < K);
        if (more) stage(cur ^ 1, k0 + BK);   // overlap refill with compute

        Frag af[2], bf[2];
        // A fragments (16x32, lane=M, two contiguous K-octets per lane-half)
        for (int mt = 0; mt < 2; ++mt) {
            int m = wm * 32 + mt * 16 + lm;
            af[mt].q[0] = *reinterpret_cast<const uint4*>(&sA[cur][m * BK + lh * 8]);
            af[mt].q[1] = *reinterpret_cast<const uint4*>(&sA[cur][m * BK + 16 + lh * 8]);
        }
        // B fragments (32x16, lane=N, two contiguous K-octets per lane-half)
        for (int nt = 0; nt < 2; ++nt) {
            int n = wn * 32 + nt * 16 + lm;
            const uint4* p = reinterpret_cast<const uint4*>(&sB[cur][n * BK + lh * 8]);
            bf[nt].q[0] = p[0];
            bf[nt].q[1] = p[2];
        }
        for (int mt = 0; mt < 2; ++mt)
            for (int nt = 0; nt < 2; ++nt)
                acc[mt][nt] = WMMA_BF16(af[mt].v, bf[nt].v, acc[mt][nt]);

        if (more) {
            wait_async_all();   // next buffer fully in LDS
            __syncthreads();    // all waves done with cur, nxt visible to all
            cur ^= 1;
        }
    }

    // Epilogue. C layout: lanes0-15 n=lane,m=r ; lanes16-31 n=lane-16,m=8+r
    for (int mt = 0; mt < 2; ++mt) {
        for (int nt = 0; nt < 2; ++nt) {
            const int nc = n0b + wn * 32 + nt * 16 + lm;
            const float bv = bias ? bias[nc] : 0.0f;
            for (int r = 0; r < 8; ++r) {
                const int mr = m0b + wm * 32 + mt * 16 + r + lh * 8;
                float v = acc[mt][nt][r] + bv;
                if (RELU) v = fmaxf(v, 0.0f);
                if (MODE == 0) {
                    ((bf16_t*)outp)[(size_t)mr * N + nc] = (bf16_t)v;
                } else if (MODE == 1) {             // [B,H,T,HS]
                    int b = mr / TT, t = mr % TT;
                    int h = nc / HS, d = nc % HS;
                    ((bf16_t*)outp)[(((size_t)(b * HH + h)) * TT + t) * HS + d] = (bf16_t)v;
                } else if (MODE == 2) {             // [B,H,HS,T]
                    int b = mr / TT, t = mr % TT;
                    int h = nc / HS, d = nc % HS;
                    ((bf16_t*)outp)[(((size_t)(b * HH + h)) * HS + d) * TT + t] = (bf16_t)v;
                } else {                            // f32 + residual
                    float rv = res ? res[(size_t)mr * N + nc] : 0.0f;
                    ((float*)outp)[(size_t)mr * N + nc] = v + rv;
                }
            }
        }
    }
}

// ---------------------------------------------------------------------------
// Causal flash attention. Grid: (T/128, B*H). 8 waves; wave owns 16 queries.
// Q,K: [B,H,T,HS] bf16. Vt: [B,H,HS,T] bf16. Out: [B,T,H*HS] bf16.
// ---------------------------------------------------------------------------
__global__ __launch_bounds__(256)
void attn_kernel(const bf16_t* __restrict__ Q,
                 const bf16_t* __restrict__ Kmat,
                 const bf16_t* __restrict__ Vt,
                 bf16_t* __restrict__ attn)
{
    __shared__ bf16_t sP[8][16 * 32];   // per-wave P scratch (C-layout -> A-layout)

    const int tid  = threadIdx.x;
    const int lane = tid & 31;
    const int wv   = tid >> 5;
    const int lm   = lane & 15;
    const int lh   = lane >> 4;
    const int bh   = blockIdx.y;
    const int q0   = blockIdx.x * 128 + wv * 16;

    const bf16_t* Qp = Q    + (size_t)bh * TT * HS;
    const bf16_t* Kp = Kmat + (size_t)bh * TT * HS;
    const bf16_t* Vp = Vt   + (size_t)bh * HS * TT;

    // Q A-fragments over HS=64 (two K-steps of 32)
    Frag aq[2];
    {
        const bf16_t* qrow = Qp + (size_t)(q0 + lm) * HS;
        for (int ks = 0; ks < 2; ++ks) {
            aq[ks].q[0] = *reinterpret_cast<const uint4*>(qrow + ks * 32 + lh * 8);
            aq[ks].q[1] = *reinterpret_cast<const uint4*>(qrow + ks * 32 + 16 + lh * 8);
        }
    }

    v8f o[4] = {};
    float mrun[8], lrun[8];
    for (int r = 0; r < 8; ++r) { mrun[r] = -1e30f; lrun[r] = 0.f; }
    const float scale = 0.125f;   // HS^-0.5

    bf16_t* pbuf = sP[wv];
    const int kend = q0 + 15;

    for (int kb = 0; kb <= kend; kb += 32) {
        // K rows act directly as B operand (lane = key, contraction over d)
        Frag bk[2][2];
        for (int nt = 0; nt < 2; ++nt) {
            const bf16_t* krow = Kp + (size_t)(kb + nt * 16 + lm) * HS;
            for (int ks = 0; ks < 2; ++ks) {
                bk[ks][nt].q[0] = *reinterpret_cast<const uint4*>(krow + ks * 32 + lh * 8);
                bk[ks][nt].q[1] = *reinterpret_cast<const uint4*>(krow + ks * 32 + 16 + lh * 8);
            }
        }
        v8f s[2] = {};
        for (int nt = 0; nt < 2; ++nt) {
            s[nt] = WMMA_BF16(aq[0].v, bk[0][nt].v, s[nt]);
            s[nt] = WMMA_BF16(aq[1].v, bk[1][nt].v, s[nt]);
        }

        // scale + causal mask
        const bool diag = (kb + 31 > q0);
        for (int nt = 0; nt < 2; ++nt)
            for (int r = 0; r < 8; ++r) {
                float v = s[nt][r] * scale;
                if (diag) {
                    int key = kb + nt * 16 + lm;
                    int qr  = q0 + r + lh * 8;
                    if (key > qr) v = -1e30f;
                }
                s[nt][r] = v;
            }

        // online softmax (row reductions stay within 16-lane halves)
        float mnew[8], alpha[8];
        for (int r = 0; r < 8; ++r) {
            float v = fmaxf(s[0][r], s[1][r]);
            for (int off = 8; off > 0; off >>= 1) v = fmaxf(v, __shfl_xor(v, off, 32));
            mnew[r]  = fmaxf(mrun[r], v);
            alpha[r] = __expf(mrun[r] - mnew[r]);
        }
        for (int nt = 0; nt < 2; ++nt)
            for (int r = 0; r < 8; ++r)
                s[nt][r] = __expf(s[nt][r] - mnew[r]);
        for (int r = 0; r < 8; ++r) {
            float v = s[0][r] + s[1][r];
            for (int off = 8; off > 0; off >>= 1) v += __shfl_xor(v, off, 32);
            lrun[r] = lrun[r] * alpha[r] + v;
            mrun[r] = mnew[r];
        }
        for (int dt = 0; dt < 4; ++dt)
            for (int r = 0; r < 8; ++r)
                o[dt][r] *= alpha[r];

        // P: C-layout -> LDS -> A-layout (wave-local, DScnt waits)
        for (int nt = 0; nt < 2; ++nt)
            for (int r = 0; r < 8; ++r)
                pbuf[(r + lh * 8) * 32 + nt * 16 + lm] = (bf16_t)s[nt][r];
        __asm__ volatile("s_wait_dscnt 0" ::: "memory");

        Frag ap;
        ap.q[0] = *reinterpret_cast<const uint4*>(&pbuf[lm * 32 + lh * 8]);
        ap.q[1] = *reinterpret_cast<const uint4*>(&pbuf[lm * 32 + 16 + lh * 8]);

        // PV: B operand rows are contiguous keys of V^T
        for (int dt = 0; dt < 4; ++dt) {
            Frag bv;
            const bf16_t* vrow = Vp + (size_t)(dt * 16 + lm) * TT + kb + lh * 8;
            bv.q[0] = *reinterpret_cast<const uint4*>(vrow);
            bv.q[1] = *reinterpret_cast<const uint4*>(vrow + 16);
            o[dt] = WMMA_BF16(ap.v, bv.v, o[dt]);
        }
        __asm__ volatile("s_wait_dscnt 0" ::: "memory");
    }

    // normalize + store to [B,T,H*HS]
    const int b = bh / HH, h = bh % HH;
    for (int r = 0; r < 8; ++r) {
        const float inv = 1.0f / lrun[r];
        const int m = q0 + r + lh * 8;
        size_t base = ((size_t)b * TT + m) * (HH * HS) + h * HS;
        for (int dt = 0; dt < 4; ++dt)
            attn[base + dt * 16 + lm] = (bf16_t)(o[dt][r] * inv);
    }
}

// ---------------------------------------------------------------------------
// Host-side orchestration
// ---------------------------------------------------------------------------
static constexpr size_t MB = 1024ull * 1024ull;
static constexpr size_t OFF_WQT  = 0;          // 2 MB
static constexpr size_t OFF_WKT  = 2 * MB;     // 2 MB
static constexpr size_t OFF_WVT  = 4 * MB;     // 2 MB
static constexpr size_t OFF_WPT  = 6 * MB;     // 2 MB
static constexpr size_t OFF_W1T  = 8 * MB;     // 8 MB
static constexpr size_t OFF_W2T  = 16 * MB;    // 8 MB
static constexpr size_t OFF_H    = 24 * MB;    // 16 MB (h1 then h2, bf16)
static constexpr size_t OFF_X1   = 40 * MB;    // 32 MB (f32)
static constexpr size_t OFF_ATTN = 72 * MB;    // 16 MB
static constexpr size_t OFF_Q    = 88 * MB;    // 16 MB
static constexpr size_t OFF_K    = 104 * MB;   // 16 MB
static constexpr size_t OFF_VT   = 120 * MB;   // 16 MB
static constexpr size_t OFF_FF1  = 88 * MB;    // 64 MB (reuses Q/K/Vt after attention)

extern "C" void kernel_launch(void* const* d_in, const int* in_sizes, int n_in,
                              void* d_out, int out_size, void* d_ws, size_t ws_size,
                              hipStream_t stream)
{
    const float* x     = (const float*)d_in[0];
    const float* Wq    = (const float*)d_in[1];
    const float* Wk    = (const float*)d_in[2];
    const float* Wv    = (const float*)d_in[3];
    const float* Wproj = (const float*)d_in[4];
    const float* bproj = (const float*)d_in[5];
    const float* W1    = (const float*)d_in[6];
    const float* b1    = (const float*)d_in[7];
    const float* W2    = (const float*)d_in[8];
    const float* b2    = (const float*)d_in[9];
    const float* g1    = (const float*)d_in[10];
    const float* be1   = (const float*)d_in[11];
    const float* g2    = (const float*)d_in[12];
    const float* be2   = (const float*)d_in[13];

    char* ws = (char*)d_ws;
    bf16_t* WqT  = (bf16_t*)(ws + OFF_WQT);
    bf16_t* WkT  = (bf16_t*)(ws + OFF_WKT);
    bf16_t* WvT  = (bf16_t*)(ws + OFF_WVT);
    bf16_t* WpT  = (bf16_t*)(ws + OFF_WPT);
    bf16_t* W1T  = (bf16_t*)(ws + OFF_W1T);
    bf16_t* W2T  = (bf16_t*)(ws + OFF_W2T);
    bf16_t* hbf  = (bf16_t*)(ws + OFF_H);
    float*  x1   = (float*) (ws + OFF_X1);
    bf16_t* atb  = (bf16_t*)(ws + OFF_ATTN);
    bf16_t* Qb   = (bf16_t*)(ws + OFF_Q);
    bf16_t* Kb   = (bf16_t*)(ws + OFF_K);
    bf16_t* Vtb  = (bf16_t*)(ws + OFF_VT);
    bf16_t* ff1  = (bf16_t*)(ws + OFF_FF1);

    // 1) weight convert + transpose -> bf16 [N][K]
    auto tp = [&](const float* in, bf16_t* out, int batch, int K, int N) {
        size_t total = (size_t)batch * K * N;
        transpose_to_bf16_kernel<<<(unsigned)((total + 255) / 256), 256, 0, stream>>>(in, out, batch, K, N);
    };
    tp(Wq, WqT, HH, CC, HS);            // [H][HS][C]  == [H*HS][C]
    tp(Wk, WkT, HH, CC, HS);
    tp(Wv, WvT, HH, CC, HS);
    tp(Wproj, WpT, 1, HH * HS, CC);     // [C][H*HS]
    tp(W1, W1T, 1, CC, 4 * CC);         // [4C][C]
    tp(W2, W2T, 1, 4 * CC, CC);         // [C][4C]

    // 2) LN1
    layernorm_bf16_kernel<<<MM, 256, 0, stream>>>(x, g1, be1, hbf);

    // 3) Q, K, V projections (scatter to attention layouts)
    dim3 gQKV(CC / 128, MM / 64);
    gemm_wmma_kernel<1, false><<<gQKV, 256, 0, stream>>>(hbf, WqT, nullptr, nullptr, Qb, MM, CC, CC);
    gemm_wmma_kernel<1, false><<<gQKV, 256, 0, stream>>>(hbf, WkT, nullptr, nullptr, Kb, MM, CC, CC);
    gemm_wmma_kernel<2, false><<<gQKV, 256, 0, stream>>>(hbf, WvT, nullptr, nullptr, Vtb, MM, CC, CC);

    // 4) causal flash attention
    attn_kernel<<<dim3(TT / 128, BB * HH), 256, 0, stream>>>(Qb, Kb, Vtb, atb);

    // 5) output projection + bias + residual(x) -> x1 (f32)
    gemm_wmma_kernel<3, false><<<dim3(CC / 128, MM / 64), 256, 0, stream>>>(atb, WpT, bproj, x, x1, MM, CC, CC);

    // 6) LN2
    layernorm_bf16_kernel<<<MM, 256, 0, stream>>>(x1, g2, be2, hbf);

    // 7) MLP: relu(h2 @ W1 + b1) -> ff1 (bf16)
    gemm_wmma_kernel<0, true><<<dim3(4 * CC / 128, MM / 64), 256, 0, stream>>>(hbf, W1T, b1, nullptr, ff1, MM, 4 * CC, CC);

    // 8) ff1 @ W2 + b2 + residual(x1) -> d_out (f32)
    gemm_wmma_kernel<3, false><<<dim3(CC / 128, MM / 64), 256, 0, stream>>>(ff1, W2T, b2, x1, (float*)d_out, MM, CC, 4 * CC);
}
